// Attention_Dec_49761491092043
// MI455X (gfx1250) — matrix operand order repaired and checked by reading, not compiled, not run on hardware
//
#include <hip/hip_runtime.h>

#define NB 8
#define NH 12
#define NS 512
#define HD 64
#define NC 768
#define LF 256
#define NBH (NB * NH)
#define SAMPLE_NUM 12800

typedef __attribute__((ext_vector_type(16))) __bf16 v16bf;
typedef __attribute__((ext_vector_type(2)))  __bf16 v2bf;
typedef __attribute__((ext_vector_type(8)))  float  v8f;
typedef __attribute__((ext_vector_type(4)))  unsigned int v4u;
typedef __attribute__((ext_vector_type(4)))  int v4i;

union BF16Frag { v16bf v; v4u u[2]; };
union F8 { v8f v; float f[8]; };

#if defined(__has_builtin)
#if __has_builtin(__builtin_amdgcn_global_load_async_to_lds_b128)
#define HAVE_ASYNC 1
#endif
#if __has_builtin(__builtin_amdgcn_cvt_pk_bf16_f32)
#define HAVE_PK_BF16 1
#endif
#endif
#ifndef HAVE_ASYNC
#define HAVE_ASYNC 0
#endif
#ifndef HAVE_PK_BF16
#define HAVE_PK_BF16 0
#endif

__device__ __forceinline__ unsigned pack2(float a, float b) {
#if HAVE_PK_BF16
  union { v2bf v; unsigned u; } r;
  r.v = __builtin_amdgcn_cvt_pk_bf16_f32(a, b);
  return r.u;
#else
  // Native __bf16 fptrunc: lets the backend use the hardware bf16 convert
  // (falls back to the software RNE expansion only if gfx1250 lacks it).
  union { v2bf v; unsigned u; } r;
  r.v.x = (__bf16)a;
  r.v.y = (__bf16)b;
  return r.u;
#endif
}
__device__ __forceinline__ unsigned short f2bf(float f) {
  union { __bf16 h; unsigned short s; } r;
  r.h = (__bf16)f;
  return r.s;
}
__device__ __forceinline__ v8f wmma_bf16(const BF16Frag& a, const BF16Frag& b, v8f c) {
  return __builtin_amdgcn_wmma_f32_16x16x32_bf16(false, a.v, false, b.v, (short)0, c,
                                                 false, false);
}
__device__ __forceinline__ void cp16_async(const void* g, void* l) {
#if HAVE_ASYNC
  __builtin_amdgcn_global_load_async_to_lds_b128(
      (__attribute__((address_space(1))) v4i*)g,
      (__attribute__((address_space(3))) v4i*)l, 0, 0);
#else
  *(v4u*)l = *(const v4u*)g;
#endif
}
__device__ __forceinline__ void wait_async() {
#if HAVE_ASYNC
  asm volatile("s_wait_asynccnt 0" ::: "memory");
#endif
}

template <int OP>  // 0=max 1=min 2=sum; blockDim.x == 256
__device__ __forceinline__ float bred(float v, float* buf) {
  const int t = threadIdx.x;
  buf[t] = v;
  __syncthreads();
  for (int s = 128; s > 0; s >>= 1) {
    if (t < s) {
      float a = buf[t], b = buf[t + s];
      buf[t] = OP == 0 ? fmaxf(a, b) : (OP == 1 ? fminf(a, b) : a + b);
    }
    __syncthreads();
  }
  float r = buf[0];
  __syncthreads();
  return r;
}

__device__ __forceinline__ unsigned hash32(unsigned a) {
  a ^= a >> 16; a *= 0x7feb352dU; a ^= a >> 15; a *= 0x846ca68bU; a ^= a >> 16;
  return a;
}

// ---------------- Kernel 1: qkv = x @ qkv_w^T  (f32 -> bf16 q/k/v, scale into q)
__global__ __launch_bounds__(256) void k_qkv(const float* __restrict__ x,
                                             const float* __restrict__ w,
                                             unsigned short* __restrict__ qB,
                                             unsigned short* __restrict__ kB,
                                             unsigned short* __restrict__ vB) {
  __shared__ __align__(16) unsigned short At[64 * 40];
  __shared__ __align__(16) unsigned short Bt[64 * 40];
  const int tid = threadIdx.x;
  const int lane = tid & 31, wv = tid >> 5;
  const int g = lane >> 4, ln = lane & 15;
  const int mSub = wv >> 1, nSub = wv & 1;
  const int rowBase = blockIdx.x * 64;
  const int oBase = blockIdx.y * 64;
  const int sr = tid >> 2, sc = (tid & 3) * 8;
  v8f acc0 = {}, acc1 = {};
  for (int kb = 0; kb < NC; kb += 32) {
    {
      const float4* sa = (const float4*)(x + (size_t)(rowBase + sr) * NC + kb + sc);
      float4 f0 = sa[0], f1 = sa[1];
      v4u pa; pa.x = pack2(f0.x, f0.y); pa.y = pack2(f0.z, f0.w);
      pa.z = pack2(f1.x, f1.y); pa.w = pack2(f1.z, f1.w);
      *(v4u*)(At + sr * 40 + sc) = pa;
      const float4* sb = (const float4*)(w + (size_t)(oBase + sr) * NC + kb + sc);
      float4 h0 = sb[0], h1 = sb[1];
      v4u pb; pb.x = pack2(h0.x, h0.y); pb.y = pack2(h0.z, h0.w);
      pb.z = pack2(h1.x, h1.y); pb.w = pack2(h1.z, h1.w);
      *(v4u*)(Bt + sr * 40 + sc) = pb;
    }
    if (kb + 32 < NC) {
      __builtin_prefetch(x + (size_t)(rowBase + sr) * NC + kb + 32 + sc, 0, 1);
      __builtin_prefetch(w + (size_t)(oBase + sr) * NC + kb + 32 + sc, 0, 1);
    }
    __syncthreads();
    BF16Frag a, b0, b1;
    a.u[0]  = *(const v4u*)(At + (mSub * 16 + ln) * 40 + g * 16);
    a.u[1]  = *(const v4u*)(At + (mSub * 16 + ln) * 40 + g * 16 + 8);
    b0.u[0] = *(const v4u*)(Bt + (nSub * 32 + ln) * 40 + g * 16);
    b0.u[1] = *(const v4u*)(Bt + (nSub * 32 + ln) * 40 + g * 16 + 8);
    b1.u[0] = *(const v4u*)(Bt + (nSub * 32 + 16 + ln) * 40 + g * 16);
    b1.u[1] = *(const v4u*)(Bt + (nSub * 32 + 16 + ln) * 40 + g * 16 + 8);
    acc0 = wmma_bf16(a, b0, acc0);
    acc1 = wmma_bf16(a, b1, acc1);
    __syncthreads();
  }
  const float scale = 0.125f;  // hd^-0.5
  F8 c0, c1; c0.v = acc0; c1.v = acc1;
  for (int rr = 0; rr < 8; ++rr) {
    int mrow = rowBase + mSub * 16 + rr + g * 8;
    int bb = mrow >> 9, nn = mrow & 511;
    for (int s = 0; s < 2; ++s) {
      int o = oBase + nSub * 32 + s * 16 + ln;
      float vl = (s == 0) ? c0.f[rr] : c1.f[rr];
      int tsel = o / NC, rem = o % NC;
      int h = rem >> 6, d = rem & 63;
      size_t dst = (((size_t)bb * NH + h) * NS + nn) * HD + d;
      if (tsel == 0) qB[dst] = f2bf(vl * scale);
      else if (tsel == 1) kB[dst] = f2bf(vl);
      else vB[dst] = f2bf(vl);
    }
  }
}

// ---------------- Kernel 2: S = q @ k^T (async LDS staging, WMMA)
__global__ __launch_bounds__(256) void k_scores(const unsigned short* __restrict__ qB,
                                                const unsigned short* __restrict__ kB,
                                                float* __restrict__ S) {
  __shared__ __align__(16) unsigned short Qt[64 * 72];
  __shared__ __align__(16) unsigned short Kt[64 * 72];
  const int tid = threadIdx.x;
  const int lane = tid & 31, wv = tid >> 5;
  const int g = lane >> 4, ln = lane & 15;
  const int mSub = wv >> 1, nSub = wv & 1;
  const int bh = blockIdx.z;
  const int iBase = blockIdx.x * 64, jBase = blockIdx.y * 64;
  const unsigned short* qb = qB + (size_t)bh * NS * HD;
  const unsigned short* kb = kB + (size_t)bh * NS * HD;
  for (int e = tid; e < 512; e += 256) {
    int row = e >> 3, ch = (e & 7) * 8;
    cp16_async(qb + (size_t)(iBase + row) * HD + ch, Qt + row * 72 + ch);
    cp16_async(kb + (size_t)(jBase + row) * HD + ch, Kt + row * 72 + ch);
  }
  wait_async();
  __syncthreads();
  v8f acc0 = {}, acc1 = {};
#pragma unroll
  for (int ks = 0; ks < 64; ks += 32) {
    BF16Frag a, b0, b1;
    a.u[0]  = *(const v4u*)(Qt + (mSub * 16 + ln) * 72 + ks + g * 16);
    a.u[1]  = *(const v4u*)(Qt + (mSub * 16 + ln) * 72 + ks + g * 16 + 8);
    b0.u[0] = *(const v4u*)(Kt + (nSub * 32 + ln) * 72 + ks + g * 16);
    b0.u[1] = *(const v4u*)(Kt + (nSub * 32 + ln) * 72 + ks + g * 16 + 8);
    b1.u[0] = *(const v4u*)(Kt + (nSub * 32 + 16 + ln) * 72 + ks + g * 16);
    b1.u[1] = *(const v4u*)(Kt + (nSub * 32 + 16 + ln) * 72 + ks + g * 16 + 8);
    acc0 = wmma_bf16(a, b0, acc0);
    acc1 = wmma_bf16(a, b1, acc1);
  }
  F8 c0, c1; c0.v = acc0; c1.v = acc1;
  float* srow = S + (size_t)bh * NS * NS;
  for (int rr = 0; rr < 8; ++rr) {
    int i = iBase + mSub * 16 + rr + g * 8;
    srow[(size_t)i * NS + jBase + nSub * 32 + ln]      = c0.f[rr];
    srow[(size_t)i * NS + jBase + nSub * 32 + 16 + ln] = c1.f[rr];
  }
}

// ---------------- Kernel 3: softmax + spatial weights + Gumbel keys + histogram
__global__ __launch_bounds__(256) void k_keys(const float* __restrict__ S,
                                              float* __restrict__ keys,
                                              unsigned* __restrict__ hist) {
  __shared__ float sm[512];
  __shared__ float red[256];
  const int t = threadIdx.x;
  const int i = blockIdx.x, f = blockIdx.y;
  const float* row = S + ((size_t)f * NS + i) * NS;
  float s0 = row[t], s1 = row[t + 256];
  float m = bred<0>(fmaxf(s0, s1), red);
  float e0 = __expf(s0 - m), e1 = __expf(s1 - m);
  sm[t] = e0; sm[t + 256] = e1;
  float denom = bred<2>(e0 + e1, red);
  const int half = i >> 8;
  const int sameBase = half << 8;
  const int crossBase = 256 - sameBase;
  float cross = sm[crossBase + t] / denom;
  float aqmax = bred<0>(cross, red) + 1e-6f;
  float a = sm[sameBase + t] / denom;
  float amin = bred<1>(a, red);
  float ap = a + fmaxf(0.f, -amin) + 1e-6f;
  if (t == i - sameBase) ap = 0.f;  // zero the diagonal
  float ssum = bred<2>(ap, red);
  float key;
  if (ap <= 0.f) {
    key = -1e30f;
  } else {
    unsigned hv = hash32(((unsigned)f * 0x9E3779B1u) ^ ((unsigned)i * 0x85EBCA77u) ^
                         ((unsigned)t * 0xC2B2AE3Du) ^ 0x27220A95u);
    float u = (float)(hv >> 8) * (1.f / 16777216.f);
    u = fminf(fmaxf(u, 1e-7f), 1.f - 1e-7f);
    float gmb = -__logf(-__logf(u));
    key = __logf(ap / ssum * aqmax) + gmb;
    key = fminf(fmaxf(key, -23.999f), 7.999f);
  }
  keys[((size_t)f * NS + i) * LF + t] = key;
  if (key > -1e29f) {
    int bin = (int)floorf((key + 24.f) * 8.f);
    bin = bin < 0 ? 0 : (bin > 255 ? 255 : bin);
    atomicAdd(&hist[f * 256 + bin], 1u);
  }
}

// ---------------- Kernel 4: per-head Gumbel-top-k threshold from histogram
__global__ void k_tau(const unsigned* __restrict__ hist, float* __restrict__ tau) {
  const int f = blockIdx.x;
  if (threadIdx.x == 0) {
    unsigned cum = 0;
    float tv = -1e30f;
    for (int b = 255; b >= 0; --b) {
      cum += hist[f * 256 + b];
      if (cum >= (unsigned)SAMPLE_NUM) { tv = -24.f + (float)b * 0.125f; break; }
    }
    tau[f] = tv;
  }
}

// ---------------- Kernel 5: masked softmax, probabilities written in-place
__global__ __launch_bounds__(256) void k_mask_softmax(float* __restrict__ S,
                                                      const float* __restrict__ keys,
                                                      const float* __restrict__ tau) {
  __shared__ float red[256];
  const int t = threadIdx.x;
  const int i = blockIdx.x, f = blockIdx.y;
  float* row = S + ((size_t)f * NS + i) * NS;
  const int half = i >> 8;
  const float tv = tau[f];
  const float* krow = keys + ((size_t)f * NS + i) * LF;
  float s0 = row[t], s1 = row[t + 256];
  bool d0, d1;
  float kk = krow[t];  // same-half column index == t for both halves
  if (half == 0) { d0 = kk > tv; d1 = false; }
  else           { d0 = false;   d1 = kk > tv; }
  float m = bred<0>(fmaxf(d0 ? -3e38f : s0, d1 ? -3e38f : s1), red);
  float e0 = d0 ? 0.f : __expf(s0 - m);
  float e1 = d1 ? 0.f : __expf(s1 - m);
  float denom = bred<2>(e0 + e1, red);
  float inv = 1.f / denom;
  row[t] = e0 * inv;
  row[t + 256] = e1 * inv;
}

// ---------------- Kernel 6: ctx = P @ v  (v transposed through LDS, WMMA)
__global__ __launch_bounds__(256) void k_av(const float* __restrict__ P,
                                            const unsigned short* __restrict__ vB,
                                            unsigned short* __restrict__ ctx) {
  __shared__ __align__(16) unsigned short Vt[64 * 40];  // [d][k] transposed
  const int tid = threadIdx.x;
  const int lane = tid & 31, wv = tid >> 5;
  const int g = lane >> 4, ln = lane & 15;
  const int bh = blockIdx.y;
  const int rowBase = blockIdx.x * 128;
  const int b = bh / NH, h = bh % NH;
  const float* pb = P + (size_t)bh * NS * NS;
  const unsigned short* vb = vB + (size_t)bh * NS * HD;
  v8f acc0 = {}, acc1 = {}, acc2 = {}, acc3 = {};
  const int skk = tid >> 3, sd0 = (tid & 7) * 8;
  for (int kb = 0; kb < NS; kb += 32) {
    union { v4u u; unsigned short s[8]; } pv;
    pv.u = *(const v4u*)(vb + (size_t)(kb + skk) * HD + sd0);
#pragma unroll
    for (int q = 0; q < 8; ++q) Vt[(sd0 + q) * 40 + skk] = pv.s[q];
    __syncthreads();
    const float* prow = pb + (size_t)(rowBase + wv * 16 + ln) * NS + kb;
    BF16Frag a;
    {
      float4 p0 = *(const float4*)(prow + g * 16);
      float4 p1 = *(const float4*)(prow + g * 16 + 4);
      a.u[0].x = pack2(p0.x, p0.y); a.u[0].y = pack2(p0.z, p0.w);
      a.u[0].z = pack2(p1.x, p1.y); a.u[0].w = pack2(p1.z, p1.w);
      float4 p2 = *(const float4*)(prow + g * 16 + 8);
      float4 p3 = *(const float4*)(prow + g * 16 + 12);
      a.u[1].x = pack2(p2.x, p2.y); a.u[1].y = pack2(p2.z, p2.w);
      a.u[1].z = pack2(p3.x, p3.y); a.u[1].w = pack2(p3.z, p3.w);
    }
    BF16Frag b0, b1, b2, b3;
    b0.u[0] = *(const v4u*)(Vt + (0 * 16 + ln) * 40 + g * 16);
    b0.u[1] = *(const v4u*)(Vt + (0 * 16 + ln) * 40 + g * 16 + 8);
    b1.u[0] = *(const v4u*)(Vt + (1 * 16 + ln) * 40 + g * 16);
    b1.u[1] = *(const v4u*)(Vt + (1 * 16 + ln) * 40 + g * 16 + 8);
    b2.u[0] = *(const v4u*)(Vt + (2 * 16 + ln) * 40 + g * 16);
    b2.u[1] = *(const v4u*)(Vt + (2 * 16 + ln) * 40 + g * 16 + 8);
    b3.u[0] = *(const v4u*)(Vt + (3 * 16 + ln) * 40 + g * 16);
    b3.u[1] = *(const v4u*)(Vt + (3 * 16 + ln) * 40 + g * 16 + 8);
    acc0 = wmma_bf16(a, b0, acc0);
    acc1 = wmma_bf16(a, b1, acc1);
    acc2 = wmma_bf16(a, b2, acc2);
    acc3 = wmma_bf16(a, b3, acc3);
    __syncthreads();
  }
  F8 c[4]; c[0].v = acc0; c[1].v = acc1; c[2].v = acc2; c[3].v = acc3;
  for (int s = 0; s < 4; ++s)
    for (int rr = 0; rr < 8; ++rr) {
      int n = rowBase + wv * 16 + rr + g * 8;
      int d = s * 16 + ln;
      ctx[((size_t)(b * NS + n)) * NC + h * HD + d] = f2bf(c[s].f[rr]);
    }
}

// ---------------- Kernel 7: out = ctx @ proj_w^T + proj_b
__global__ __launch_bounds__(256) void k_proj(const unsigned short* __restrict__ ctx,
                                              const float* __restrict__ w,
                                              const float* __restrict__ bias,
                                              float* __restrict__ out) {
  __shared__ __align__(16) unsigned short At[64 * 40];
  __shared__ __align__(16) unsigned short Bt[64 * 40];
  const int tid = threadIdx.x;
  const int lane = tid & 31, wv = tid >> 5;
  const int g = lane >> 4, ln = lane & 15;
  const int mSub = wv >> 1, nSub = wv & 1;
  const int rowBase = blockIdx.x * 64;
  const int colBase = blockIdx.y * 64;
  const int ar = tid >> 2, ac = (tid & 3) * 8;
  v8f acc0 = {}, acc1 = {};
  for (int kb = 0; kb < NC; kb += 32) {
    cp16_async(ctx + (size_t)(rowBase + ar) * NC + kb + ac, At + ar * 40 + ac);
    {
      const float4* sb = (const float4*)(w + (size_t)(colBase + ar) * NC + kb + ac);
      float4 h0 = sb[0], h1 = sb[1];
      v4u pbk; pbk.x = pack2(h0.x, h0.y); pbk.y = pack2(h0.z, h0.w);
      pbk.z = pack2(h1.x, h1.y); pbk.w = pack2(h1.z, h1.w);
      *(v4u*)(Bt + ar * 40 + ac) = pbk;
    }
    if (kb + 32 < NC)
      __builtin_prefetch(w + (size_t)(colBase + ar) * NC + kb + 32 + ac, 0, 1);
    wait_async();
    __syncthreads();
    BF16Frag a, b0, b1;
    a.u[0]  = *(const v4u*)(At + (mSub * 16 + ln) * 40 + g * 16);
    a.u[1]  = *(const v4u*)(At + (mSub * 16 + ln) * 40 + g * 16 + 8);
    b0.u[0] = *(const v4u*)(Bt + (nSub * 32 + ln) * 40 + g * 16);
    b0.u[1] = *(const v4u*)(Bt + (nSub * 32 + ln) * 40 + g * 16 + 8);
    b1.u[0] = *(const v4u*)(Bt + (nSub * 32 + 16 + ln) * 40 + g * 16);
    b1.u[1] = *(const v4u*)(Bt + (nSub * 32 + 16 + ln) * 40 + g * 16 + 8);
    acc0 = wmma_bf16(a, b0, acc0);
    acc1 = wmma_bf16(a, b1, acc1);
    __syncthreads();
  }
  F8 c0, c1; c0.v = acc0; c1.v = acc1;
  for (int rr = 0; rr < 8; ++rr) {
    int mrow = rowBase + mSub * 16 + rr + g * 8;
    int col0 = colBase + nSub * 32 + ln;
    out[(size_t)mrow * NC + col0]      = c0.f[rr] + bias[col0];
    out[(size_t)mrow * NC + col0 + 16] = c1.f[rr] + bias[col0 + 16];
  }
}

extern "C" void kernel_launch(void* const* d_in, const int* in_sizes, int n_in,
                              void* d_out, int out_size, void* d_ws, size_t ws_size,
                              hipStream_t stream) {
  (void)in_sizes; (void)n_in; (void)out_size; (void)ws_size;
  const float* x      = (const float*)d_in[0];
  // d_in[1] (mask) is unused by the reference forward pass.
  const float* qkv_w  = (const float*)d_in[2];
  const float* proj_w = (const float*)d_in[3];
  const float* proj_b = (const float*)d_in[4];
  float* out = (float*)d_out;

  char* ws = (char*)d_ws;
  size_t off = 0;
  auto carve = [&](size_t bytes) -> char* {
    char* p = ws + off;
    off += (bytes + 255) & ~(size_t)255;
    return p;
  };
  unsigned short* qB  = (unsigned short*)carve((size_t)NBH * NS * HD * 2);
  unsigned short* kB  = (unsigned short*)carve((size_t)NBH * NS * HD * 2);
  unsigned short* vB  = (unsigned short*)carve((size_t)NBH * NS * HD * 2);
  float* S            = (float*)carve((size_t)NBH * NS * NS * 4);
  float* keys         = (float*)carve((size_t)NBH * NS * LF * 4);
  unsigned short* ctx = (unsigned short*)carve((size_t)NB * NS * NC * 2);
  unsigned* hist      = (unsigned*)carve((size_t)NBH * 256 * 4);
  float* tau          = (float*)carve((size_t)NBH * 4);

  (void)hipMemsetAsync(hist, 0, (size_t)NBH * 256 * 4, stream);
  k_qkv<<<dim3(64, 36), 256, 0, stream>>>(x, qkv_w, qB, kB, vB);
  k_scores<<<dim3(8, 8, 96), 256, 0, stream>>>(qB, kB, S);
  k_keys<<<dim3(512, 96), 256, 0, stream>>>(S, keys, hist);
  k_tau<<<dim3(96), dim3(32), 0, stream>>>(hist, tau);
  k_mask_softmax<<<dim3(512, 96), 256, 0, stream>>>(S, keys, tau);
  k_av<<<dim3(4, 96), 256, 0, stream>>>(S, vB, ctx);
  k_proj<<<dim3(64, 12), 256, 0, stream>>>(ctx, proj_w, proj_b, out);
}
